// RNNTensorflowModelImpl_39256001085984
// MI455X (gfx1250) — matrix operand order repaired and checked by reading, not compiled, not run on hardware
//
#include <hip/hip_runtime.h>
#include <hip/hip_bf16.h>

typedef __attribute__((ext_vector_type(16))) __bf16 v16bf;
typedef __attribute__((ext_vector_type(8)))  float  v8f;

// ---------------- constants for this problem ----------------
#define BB 32
#define TT 256
#define EE 2048
#define HH 256
#define CC 101
#define N4H 1024           // 4*H
#define MBIG (TT*BB)       // 8192 rows of the input-projection GEMM
#define KT_BIG (EE/32)     // 64 k-tiles
#define MT_BIG (MBIG/16)   // 512 m-tiles
#define NT_BIG (N4H/16)    // 64 n-tiles
#define KFIN (TT*2*HH)     // 131072
#define KT_FIN (KFIN/32)   // 4096
#define NPAD_FIN 112       // 101 padded to 7 n-tiles
#define NT_FIN 7

// workspace byte offsets (all 1KB aligned)
#define OFF_APACK 0ll                               // 32 MB: packed x  (bf16 A-tiles)
#define OFF_BF    33554432ll                        //  4 MB: packed kernel_fw[0:E]
#define OFF_BBW   37748736ll                        //  4 MB: packed kernel_bw[0:E]
#define OFF_WHF   41943040ll                        // 512KB: packed kernel_fw[E:E+H]
#define OFF_WHB   42467328ll                        // 512KB: packed kernel_bw[E:E+H]
#define OFF_ZX    42991616ll                        // 64 MB: Zx fw then bw, fp32
#define OFF_HCAT  110100480ll                       //  8 MB: packed tanh(h) bf16 A-tiles
#define OFF_WPK   118489088ll                       // 28 MB: packed w bf16 B-tiles

// ---------------- helpers ----------------
__device__ __forceinline__ __bf16 f2bf(float f) {
    unsigned u = __float_as_uint(f);
    u += 0x7fffu + ((u >> 16) & 1u);                // round to nearest even
    unsigned short s = (unsigned short)(u >> 16);
    return __builtin_bit_cast(__bf16, s);
}
__device__ __forceinline__ float sigf(float x) {
    return 1.0f / (1.0f + __expf(-x));
}
// byte offset of element (m_local, k_local) inside a 1024B 16x32 bf16 A-tile
__device__ __forceinline__ int a_slot(int ml, int kl) {
    int g    = (kl >> 3) & 1;
    int lane = ml + (g << 4);
    int j    = ((kl >> 4) << 2) + ((kl & 7) >> 1);
    return lane * 32 + j * 4 + (kl & 1) * 2;
}
// byte offset of element (k_local, n_local) inside a 1024B 32x16 bf16 B-tile
__device__ __forceinline__ int b_slot(int kl, int nl) {
    int lane = nl + (((kl >> 3) & 1) << 4);
    int j    = ((kl >> 4) << 2) + ((kl & 7) >> 1);
    return lane * 32 + j * 4 + (kl & 1) * 2;
}
__device__ __forceinline__ v8f wmma_bf16(v16bf a, v16bf b, v8f c) {
    return __builtin_amdgcn_wmma_f32_16x16x32_bf16(false, a, false, b, (short)0, c,
                                                   false, false);
}

// ---- CDNA5 async global->LDS copy via inline asm (ASYNCcnt-tracked) ----
// LDS byte address = low 32 bits of the generic pointer (flat-aperture rule).
__device__ __forceinline__ void async_cp16(const void* g, void* l) {
    unsigned           lds  = (unsigned)(unsigned long long)l;
    unsigned long long gadr = (unsigned long long)g;
    asm volatile("global_load_async_to_lds_b128 %0, %1, off"
                 :: "v"(lds), "v"(gadr) : "memory");
}
__device__ __forceinline__ void wait_async0() {
    asm volatile("s_wait_asynccnt 0x0" ::: "memory");
}
__device__ __forceinline__ void wait_async4() {
    asm volatile("s_wait_asynccnt 0x4" ::: "memory");
}

// ---------------- pack kernels ----------------
// A-pack of x: row m = t*32+b, col k ; A[m][k] = x[b][t][k]
__global__ void pack_x_kernel(const float* __restrict__ x, char* __restrict__ out) {
    long long idx = (long long)blockIdx.x * blockDim.x + threadIdx.x;
    if (idx >= (long long)MBIG * EE) return;
    int k = (int)(idx % EE);
    long long m = idx / EE;
    int b = (int)(m & 31), t = (int)(m >> 5);
    float v = x[((long long)b * TT + t) * EE + k];
    long long tile = (long long)(m >> 4) * KT_BIG + (k >> 5);
    *(__bf16*)(out + tile * 1024 + a_slot((int)(m & 15), k & 31)) = f2bf(v);
}
// generic B-pack: src fp32 [rows x ld], rows row0..row0+K-1, cols 0..N-1 (pad to Npad)
__global__ void pack_b_kernel(const float* __restrict__ src, char* __restrict__ out,
                              int K, int N, int Npad, int ld, int row0) {
    long long idx = (long long)blockIdx.x * blockDim.x + threadIdx.x;
    if (idx >= (long long)K * Npad) return;
    int n = (int)(idx % Npad);
    int k = (int)(idx / Npad);
    float v = (n < N) ? src[(long long)(row0 + k) * ld + n] : 0.0f;
    long long tile = (long long)(k >> 5) * (Npad >> 4) + (n >> 4);
    *(__bf16*)(out + tile * 1024 + b_slot(k & 31, n & 15)) = f2bf(v);
}

// ---------------- big GEMM: Zx = x @ W_x + bias (LDS-staged, async double-buffer) ----
// Workgroup = 256 thr (8 waves) computes a 128x128 block; each wave a 2x4 tile grid.
// Per k-step: stage 8KB of A + 8KB of B in LDS (4 async b128 per thread), 8 WMMA/wave.
__global__ __launch_bounds__(256) void gemm_zx_tiled(const char* __restrict__ Apack,
                                                     const char* __restrict__ Bpack,
                                                     const float* __restrict__ bias,
                                                     float* __restrict__ Z) {
    __shared__ __align__(16) char sA[2][8192];
    __shared__ __align__(16) char sB[2][8192];
    int tid = threadIdx.x, lane = tid & 31, wave = tid >> 5;
    int mb = blockIdx.x;                 // 0..63
    int nb = blockIdx.y;                 // 0..7
    int wm = (wave & 3) * 2;             // wave's m-tile offset in block
    int wn = (wave >> 2) * 4;            // wave's n-tile offset in block
    const char* Ab = Apack + (long long)(mb * 8) * KT_BIG * 1024;
    const char* Bb = Bpack + (long long)(nb * 8) * 1024;

    v8f acc[8] = {};
    // preload kt = 0
    #pragma unroll
    for (int p = 0; p < 2; ++p) {
        int e = tid + p * 256;
        int tA = e >> 6, oA = (e & 63) << 4;
        async_cp16(Ab + (long long)tA * KT_BIG * 1024 + oA, &sA[0][tA * 1024 + oA]);
        async_cp16(Bb + 0ll + e * 16, &sB[0][e * 16]);
    }
    for (int kt = 0; kt < KT_BIG; ++kt) {
        int cur = kt & 1;
        if (kt + 1 < KT_BIG) {
            int nk = kt + 1, nxt = cur ^ 1;
            #pragma unroll
            for (int p = 0; p < 2; ++p) {
                int e = tid + p * 256;
                int tA = e >> 6, oA = (e & 63) << 4;
                async_cp16(Ab + ((long long)tA * KT_BIG + nk) * 1024 + oA,
                           &sA[nxt][tA * 1024 + oA]);
                async_cp16(Bb + (long long)nk * NT_BIG * 1024 + e * 16, &sB[nxt][e * 16]);
            }
            wait_async4();               // current stage (4 older ops) complete
        } else {
            wait_async0();
        }
        __syncthreads();                 // stage visible to all waves
        v16bf a0 = *(const v16bf*)&sA[cur][(wm + 0) * 1024 + lane * 32];
        v16bf a1 = *(const v16bf*)&sA[cur][(wm + 1) * 1024 + lane * 32];
        #pragma unroll
        for (int j = 0; j < 4; ++j) {
            v16bf b = *(const v16bf*)&sB[cur][(wn + j) * 1024 + lane * 32];
            acc[j]     = wmma_bf16(a0, b, acc[j]);
            acc[4 + j] = wmma_bf16(a1, b, acc[4 + j]);
        }
        __syncthreads();                 // everyone done before next-buffer overwrite
    }
    int nl = lane & 15, mh = (lane >> 4) * 8;
    #pragma unroll
    for (int mi = 0; mi < 2; ++mi) {
        #pragma unroll
        for (int j = 0; j < 4; ++j) {
            int n = (nb * 8 + wn + j) * 16 + nl;
            float bv = bias[n];
            v8f a = acc[mi * 4 + j];
            #pragma unroll
            for (int r = 0; r < 8; ++r) {
                int m = (mb * 8 + wm + mi) * 16 + mh + r;   // m = t*32 + b
                Z[(long long)m * N4H + n] = a[r] + bv;
            }
        }
    }
}

// ---------------- persistent bidirectional LSTM recurrence ----------------
// grid = 2 (fw/bw), 1024 threads = 32 waves. Per step: async-stage Zx[t] (128KB)
// into LDS overlapped with the 128-tile recurrent WMMA GEMM, fuse, gates, repack h.
__global__ __launch_bounds__(1024) void lstm_kernel(const float* __restrict__ zx_base,
                                                    const char* __restrict__ whf,
                                                    const char* __restrict__ whb,
                                                    char* __restrict__ hcat) {
    __shared__ __align__(16) float zxl[BB * N4H];         // 128 KB: Zx[t] then z
    __shared__ float cst[BB * HH];                        //  32 KB: cell state
    __shared__ __align__(32) char hpack[16 * 1024];       //  16 KB: h as A-tiles

    int dir = blockIdx.x;
    const float* Zx = zx_base + (long long)dir * TT * BB * N4H;
    const char*  Wh = dir ? whb : whf;
    int tid = threadIdx.x, lane = tid & 31, wave = tid >> 5;

    for (int i = tid; i < BB * HH; i += 1024) cst[i] = 0.0f;
    for (int i = tid; i < (16 * 1024) / 4; i += 1024) ((int*)hpack)[i] = 0;
    __syncthreads();

    for (int s = 0; s < TT; ++s) {
        int t = dir ? (TT - 1 - s) : s;
        // ---- overlap: async-load Zx[t] (contiguous 128KB) into LDS ----
        const char* zsrc = (const char*)Zx + (long long)t * BB * N4H * 4;
        #pragma unroll
        for (int p = 0; p < 8; ++p)
            async_cp16(zsrc + tid * 16 + p * 16384, (char*)zxl + tid * 16 + p * 16384);
        // ---- recurrent GEMM: acc = h_prev @ W_h  (4 tiles per wave) ----
        v8f acc[4];
        #pragma unroll
        for (int tt = 0; tt < 4; ++tt) {
            int idx = wave * 4 + tt;
            int mt = idx >> 6, nt = idx & 63;
            v8f a = {};
            #pragma unroll
            for (int kt = 0; kt < 8; ++kt) {
                v16bf av = *(const v16bf*)(hpack + (mt * 8 + kt) * 1024 + lane * 32);
                v16bf bv = *(const v16bf*)(Wh + (long long)(kt * 64 + nt) * 1024 + lane * 32);
                a = wmma_bf16(av, bv, a);
            }
            acc[tt] = a;
        }
        wait_async0();
        __syncthreads();                 // zxl ready, hpack reads done
        // ---- fuse: z = Zx[t] + h@Wh (in place in LDS) ----
        #pragma unroll
        for (int tt = 0; tt < 4; ++tt) {
            int idx = wave * 4 + tt;
            int mt = idx >> 6, nt = idx & 63;
            int nl = lane & 15, mh = (lane >> 4) * 8;
            int n  = nt * 16 + nl;
            #pragma unroll
            for (int r = 0; r < 8; ++r) {
                int bi = mt * 16 + mh + r;
                zxl[bi * N4H + n] += acc[tt][r];
            }
        }
        __syncthreads();
        // ---- gates + state update (sigmoid activation per reference) ----
        #pragma unroll
        for (int e = 0; e < 8; ++e) {
            int idx = tid * 8 + e;                 // 8192 = 32*256 elements
            int bi = idx >> 8, j = idx & 255;
            const float* zr = &zxl[bi * N4H];
            float gi = sigf(zr[j]);
            float gj = sigf(zr[256 + j]);
            float gf = sigf(zr[512 + j] + 1.0f);   // FORGET_BIAS
            float go = sigf(zr[768 + j]);
            float c  = gf * cst[bi * HH + j] + gi * gj;
            cst[bi * HH + j] = c;
            float h = go * sigf(c);                // activation=sigmoid, not tanh
            *(__bf16*)(hpack + ((bi >> 4) * 8 + (j >> 5)) * 1024 +
                       a_slot(bi & 15, j & 31)) = f2bf(h);
            int kk = t * 512 + dir * 256 + j;      // col in [B, T*2H]
            long long tile = (long long)(bi >> 4) * KT_FIN + (kk >> 5);
            *(__bf16*)(hcat + tile * 1024 + a_slot(bi & 15, kk & 31)) = f2bf(tanhf(h));
        }
        __syncthreads();
    }
}

// ---------------- output init + final split-K GEMM ----------------
__global__ void init_out_kernel(const float* __restrict__ bias, float* __restrict__ out) {
    int i = blockIdx.x * blockDim.x + threadIdx.x;
    if (i < BB * CC) out[i] = bias[i % CC];
}
// 14 C-tiles x 64 K-splits = 896 waves; 64 k-tiles each; atomic-add partials.
__global__ void gemm_final_kernel(const char* __restrict__ Apack,
                                  const char* __restrict__ Bpack,
                                  float* __restrict__ out) {
    int gwave = (int)((blockIdx.x * (long long)blockDim.x + threadIdx.x) >> 5);
    int lane  = threadIdx.x & 31;
    int split = gwave & 63;
    int tile  = gwave >> 6;                  // 0..13
    int nt = tile % NT_FIN, mt = tile / NT_FIN;
    int k0 = split * 64;
    const char* ab = Apack + ((long long)mt * KT_FIN + k0) * 1024 + lane * 32;
    const char* bb = Bpack + ((long long)k0 * NT_FIN + nt) * 1024 + lane * 32;
    v8f acc = {};
    for (int kt = 0; kt < 64; ++kt) {
        v16bf a = *(const v16bf*)(ab + (long long)kt * 1024);
        v16bf b = *(const v16bf*)(bb + (long long)kt * NT_FIN * 1024);
        acc = wmma_bf16(a, b, acc);
    }
    int nl = lane & 15, mh = (lane >> 4) * 8;
    int n  = nt * 16 + nl;
    if (n < CC) {
        #pragma unroll
        for (int r = 0; r < 8; ++r) {
            int m = mt * 16 + mh + r;
            atomicAdd(&out[m * CC + n], acc[r]);
        }
    }
}

// ---------------- host launch ----------------
extern "C" void kernel_launch(void* const* d_in, const int* in_sizes, int n_in,
                              void* d_out, int out_size, void* d_ws, size_t ws_size,
                              hipStream_t stream) {
    const float* x         = (const float*)d_in[0];
    const float* kernel_fw = (const float*)d_in[1];
    const float* bias_fw   = (const float*)d_in[2];
    const float* kernel_bw = (const float*)d_in[3];
    const float* bias_bw   = (const float*)d_in[4];
    const float* w         = (const float*)d_in[5];
    const float* bvec      = (const float*)d_in[6];
    float* out = (float*)d_out;
    char*  ws  = (char*)d_ws;

    char* apack = ws + OFF_APACK;
    char* bpf   = ws + OFF_BF;
    char* bpb   = ws + OFF_BBW;
    char* whf   = ws + OFF_WHF;
    char* whb   = ws + OFF_WHB;
    float* zx   = (float*)(ws + OFF_ZX);           // fw then bw
    char* hcat  = ws + OFF_HCAT;
    char* wpk   = ws + OFF_WPK;

    // 1) pack operands to bf16 WMMA tile layouts
    pack_x_kernel<<<(MBIG * (long long)EE + 255) / 256, 256, 0, stream>>>(x, apack);
    pack_b_kernel<<<(EE * N4H + 255) / 256, 256, 0, stream>>>(kernel_fw, bpf, EE, N4H, N4H, N4H, 0);
    pack_b_kernel<<<(EE * N4H + 255) / 256, 256, 0, stream>>>(kernel_bw, bpb, EE, N4H, N4H, N4H, 0);
    pack_b_kernel<<<(HH * N4H + 255) / 256, 256, 0, stream>>>(kernel_fw, whf, HH, N4H, N4H, N4H, EE);
    pack_b_kernel<<<(HH * N4H + 255) / 256, 256, 0, stream>>>(kernel_bw, whb, HH, N4H, N4H, N4H, EE);
    pack_b_kernel<<<((long long)KFIN * NPAD_FIN + 255) / 256, 256, 0, stream>>>(w, wpk, KFIN, CC, NPAD_FIN, CC, 0);

    // 2) input projections (time-parallel, LDS-staged bf16 WMMA)
    dim3 zgrid(MT_BIG / 8, NT_BIG / 8);            // 64 x 8 workgroups
    gemm_zx_tiled<<<zgrid, 256, 0, stream>>>(apack, bpf, bias_fw, zx);
    gemm_zx_tiled<<<zgrid, 256, 0, stream>>>(apack, bpb, bias_bw, zx + (long long)TT * BB * N4H);

    // 3) output init with bias (final GEMM atomically accumulates)
    init_out_kernel<<<(BB * CC + 255) / 256, 256, 0, stream>>>(bvec, out);

    // 4) sequential recurrence: 1 persistent WG per direction
    lstm_kernel<<<2, 1024, 0, stream>>>(zx, whf, whb, hcat);

    // 5) final projection, split-K with f32 atomics
    gemm_final_kernel<<<(14 * 64) / 8, 256, 0, stream>>>(hcat, wpk, out);
}
